// NeuralAttention_50182397887017
// MI455X (gfx1250) — matrix operand order repaired and checked by reading, not compiled, not run on hardware
//
#include <hip/hip_runtime.h>
#include <hip/hip_bf16.h>

typedef _Float16 half8 __attribute__((ext_vector_type(8)));
typedef _Float16 v16h  __attribute__((ext_vector_type(16)));
typedef float    v8f   __attribute__((ext_vector_type(8)));

#define TSEQ   4096
#define HID    512
#define NHEAD  8
#define HDIM   64

__device__ __forceinline__ v16h mk16(half8 a, half8 b) {
    v16h r;
#pragma unroll
    for (int i = 0; i < 8; ++i) { r[i] = a[i]; r[i + 8] = b[i]; }
    return r;
}

__device__ __forceinline__ v8f wmma16(v16h a, v16h b, v8f c) {
    return __builtin_amdgcn_wmma_f32_16x16x32_f16(false, a, false, b, (short)0, c, false, false);
}

__device__ __forceinline__ v8f zero8() {
    v8f z;
#pragma unroll
    for (int i = 0; i < 8; ++i) z[i] = 0.0f;
    return z;
}

// gfx1250 async global->LDS copy (ASYNCcnt-tracked, no VGPR staging).
// lds_off: LDS byte offset (flat shared ptr truncated: LDS aperture keeps offset in addr[31:0]).
__device__ __forceinline__ void async_copy_b128(const void* lds_ptr, const void* gptr) {
    unsigned lds_off = (unsigned)(size_t)lds_ptr;
    asm volatile("global_load_async_to_lds_b128 %0, %1, off"
                 :: "v"(lds_off), "v"(gptr) : "memory");
}
__device__ __forceinline__ void wait_async0() {
    asm volatile("s_wait_asynccnt 0x0" ::: "memory");
}

// ---------------- prep kernels ----------------

__global__ void cvt_f32_to_f16_kernel(const float* __restrict__ src,
                                      _Float16* __restrict__ dst, int n) {
    int i = blockIdx.x * blockDim.x + threadIdx.x;
    if (i < n) dst[i] = (_Float16)src[i];
}

__global__ void rope_table_kernel(const int* __restrict__ ts,
                                  float* __restrict__ cosT,
                                  float* __restrict__ sinT) {
    int i = blockIdx.x * blockDim.x + threadIdx.x;   // 4096*32
    int t = i >> 5, f = i & 31;
    float invf = __expf(-logf(10000.0f) * (float)f * (1.0f / 32.0f));
    float ang  = (float)ts[t] * invf;
    cosT[i] = cosf(ang);
    sinT[i] = sinf(ang);
}

// ---------------- fused QKV projection + RoPE ----------------
// grid (T/128, 8, 3); block 256 (8 waves). N-tile 64 == one head.
// Q/K stored [head][t][64] with RoPE applied; V stored TRANSPOSED [head][d][t].
__global__ __launch_bounds__(256) void qkv_rope_kernel(
    const float* __restrict__ x,
    const _Float16* __restrict__ wqh, const _Float16* __restrict__ wkh,
    const _Float16* __restrict__ wvh,
    const float* __restrict__ cosT, const float* __restrict__ sinT,
    _Float16* __restrict__ qh, _Float16* __restrict__ kh, _Float16* __restrict__ vt)
{
    __shared__ _Float16 xs[128][80];

    const int t0 = blockIdx.x * 128;
    const int head = blockIdx.y;
    const int n0 = head * 64;
    const int which = blockIdx.z;
    const _Float16* W = (which == 0) ? wqh : (which == 1) ? wkh : wvh;

    const int tid = threadIdx.x;
    const int l = tid & 31, w = tid >> 5;
    const int lo = l & 15, hi = l >> 4;
    const int m0 = w * 16;

    v8f acc[4];
#pragma unroll
    for (int nt = 0; nt < 4; ++nt) acc[nt] = zero8();

    for (int kc = 0; kc < HID; kc += 64) {
        // cooperative load + f32->f16 convert of X chunk [128 x 64]
        {
            const int r = tid >> 1, c0 = (tid & 1) * 32;
            const float* src = x + (size_t)(t0 + r) * HID + kc + c0;
#pragma unroll
            for (int i = 0; i < 4; ++i) {
                half8 h;
#pragma unroll
                for (int e = 0; e < 8; ++e) h[e] = (_Float16)src[i * 8 + e];
                *(half8*)&xs[r][c0 + i * 8] = h;
            }
        }
        __syncthreads();
#pragma unroll
        for (int ks = 0; ks < 64; ks += 32) {
            const _Float16* arow = &xs[m0 + lo][ks + hi * 8];
            v16h a = mk16(*(const half8*)arow, *(const half8*)(arow + 16));
            // batch all 4 B-fragment loads so LOADcnt waits stagger (0x3..0x0)
            v16h b[4];
#pragma unroll
            for (int nt = 0; nt < 4; ++nt) {
                const _Float16* brow = W + (size_t)(n0 + nt * 16 + lo) * HID + kc + ks + hi * 8;
                b[nt] = mk16(*(const half8*)brow, *(const half8*)(brow + 16));
            }
#pragma unroll
            for (int nt = 0; nt < 4; ++nt)
                acc[nt] = wmma16(a, b[nt], acc[nt]);
        }
        __syncthreads();
    }

    if (which == 2) {
        // V transposed: vt[head][d][t]; fragment element v -> consecutive t => packed b128 stores
#pragma unroll
        for (int nt = 0; nt < 4; ++nt) {
            int d = nt * 16 + lo;
            half8 pk;
#pragma unroll
            for (int v = 0; v < 8; ++v) pk[v] = (_Float16)acc[nt][v];
            *(half8*)(vt + ((size_t)(head * HDIM + d)) * TSEQ + t0 + m0 + 8 * hi) = pk;
        }
    } else {
        _Float16* OUT = (which == 0) ? qh : kh;
        // RoPE: pair fragment nt with nt+2 (dims d and d+32 share a lane)
#pragma unroll
        for (int pr = 0; pr < 2; ++pr)
#pragma unroll
            for (int v = 0; v < 8; ++v) {
                int t = t0 + m0 + v + 8 * hi;
                int d = pr * 16 + lo;                 // 0..31
                float c = cosT[t * 32 + d];
                float s = sinT[t * 32 + d];
                float xl = acc[pr][v], xh2 = acc[pr + 2][v];
                size_t base = ((size_t)head * TSEQ + t) * HDIM;
                OUT[base + d]      = (_Float16)(xl * c - xh2 * s);
                OUT[base + d + 32] = (_Float16)(xh2 * c + xl * s);
            }
    }
}

// ---------------- flash attention ----------------
// grid (T/64, 8); block 128 (4 waves), each wave owns 16 query rows.
// K/V tiles double-buffered in LDS, filled by async global->LDS copies.
__global__ __launch_bounds__(128) void flash_attn_kernel(
    const _Float16* __restrict__ qh, const _Float16* __restrict__ kh,
    const _Float16* __restrict__ vt, _Float16* __restrict__ ah)
{
    __shared__ _Float16 ksT[2][64][80];   // K tile, row-major [key][dim]
    __shared__ _Float16 vsT[2][64][80];   // V tile, [dim][key] (pre-transposed in global)
    __shared__ _Float16 psT[4][16][80];   // per-wave P strip [qrow][key]

    const int t0 = blockIdx.x * 64;
    const int head = blockIdx.y;
    const int tid = threadIdx.x;
    const int l = tid & 31, w = tid >> 5;
    const int lo = l & 15, hi = l >> 4;
    const int m0 = w * 16;

    // stage one 64-key K/V tile into LDS buffer `buf` with async copies
    auto stage = [&](int buf, int j0) {
#pragma unroll
        for (int i = 0; i < 4; ++i) {
            int idx = i * 128 + tid;          // 512 chunks of 16B per tile
            int row = idx >> 3, ch = (idx & 7) * 8;
            async_copy_b128(&ksT[buf][row][ch],
                            kh + ((size_t)head * TSEQ + j0 + row) * HDIM + ch);
            async_copy_b128(&vsT[buf][row][ch],
                            vt + ((size_t)(head * HDIM + row)) * TSEQ + j0 + ch);
        }
    };

    // Q fragments, pre-scaled by 1/sqrt(64)
    const _Float16* qrow = qh + ((size_t)head * TSEQ + t0 + m0 + lo) * HDIM;
    v16h qa[2];
#pragma unroll
    for (int kk = 0; kk < 2; ++kk) {
        half8 a0 = *(const half8*)(qrow + kk * 32 + hi * 8);
        half8 a1 = *(const half8*)(qrow + kk * 32 + 16 + hi * 8);
#pragma unroll
        for (int e = 0; e < 8; ++e) { a0[e] = a0[e] * (_Float16)0.125f; a1[e] = a1[e] * (_Float16)0.125f; }
        qa[kk] = mk16(a0, a1);
    }

    float m_i[8], l_i[8];
    v8f o[4];
#pragma unroll
    for (int v = 0; v < 8; ++v) { m_i[v] = -3.0e38f; l_i[v] = 0.0f; }
#pragma unroll
    for (int nt = 0; nt < 4; ++nt) o[nt] = zero8();

    stage(0, 0);
    int buf = 0;
    for (int j0 = 0; j0 < TSEQ; j0 += 64) {
        wait_async0();        // my async copies for `buf` have landed
        __syncthreads();      // everyone's have landed; prev buffer fully consumed
        if (j0 + 64 < TSEQ) stage(buf ^ 1, j0 + 64);   // overlap next tile with compute

        // S = (Q*scale) . K^T  -> 16 x 64, f32
        v8f sacc[4];
#pragma unroll
        for (int kt = 0; kt < 4; ++kt) {
            sacc[kt] = zero8();
            v16h bk[2];
#pragma unroll
            for (int kk = 0; kk < 2; ++kk) {
                const _Float16* brow = &ksT[buf][kt * 16 + lo][kk * 32 + hi * 8];
                bk[kk] = mk16(*(const half8*)brow, *(const half8*)(brow + 16));
            }
#pragma unroll
            for (int kk = 0; kk < 2; ++kk)
                sacc[kt] = wmma16(qa[kk], bk[kk], sacc[kt]);
        }

        // online softmax: row max / rescale (rows live in lane groups of 16)
        float alpha[8];
#pragma unroll
        for (int v = 0; v < 8; ++v) {
            float rm = fmaxf(fmaxf(sacc[0][v], sacc[1][v]), fmaxf(sacc[2][v], sacc[3][v]));
#pragma unroll
            for (int off = 1; off < 16; off <<= 1)
                rm = fmaxf(rm, __shfl_xor(rm, off, 16));
            float mnew = fmaxf(m_i[v], rm);
            alpha[v] = __expf(m_i[v] - mnew);
            m_i[v] = mnew;
        }

        float rsum[8];
#pragma unroll
        for (int v = 0; v < 8; ++v) rsum[v] = 0.0f;
#pragma unroll
        for (int kt = 0; kt < 4; ++kt)
#pragma unroll
            for (int v = 0; v < 8; ++v) {
                float p = __expf(sacc[kt][v] - m_i[v]);
                rsum[v] += p;
                psT[w][v + 8 * hi][kt * 16 + lo] = (_Float16)p;   // relayout via LDS
            }
#pragma unroll
        for (int v = 0; v < 8; ++v) {
            float rs = rsum[v];
#pragma unroll
            for (int off = 1; off < 16; off <<= 1)
                rs += __shfl_xor(rs, off, 16);
            l_i[v] = l_i[v] * alpha[v] + rs;
#pragma unroll
            for (int nt = 0; nt < 4; ++nt) o[nt][v] *= alpha[v];
        }

        // read P back as A-fragments (wave-private strip; DS ops are in-order per wave)
        const _Float16* prow = &psT[w][lo][0];
        v16h pa[2];
#pragma unroll
        for (int kk = 0; kk < 2; ++kk)
            pa[kk] = mk16(*(const half8*)(prow + kk * 32 + hi * 8),
                          *(const half8*)(prow + kk * 32 + 16 + hi * 8));

        // O += P . V
#pragma unroll
        for (int nt = 0; nt < 4; ++nt) {
            v16h bv[2];
#pragma unroll
            for (int kk = 0; kk < 2; ++kk) {
                const _Float16* brow = &vsT[buf][nt * 16 + lo][kk * 32 + hi * 8];
                bv[kk] = mk16(*(const half8*)brow, *(const half8*)(brow + 16));
            }
#pragma unroll
            for (int kk = 0; kk < 2; ++kk)
                o[nt] = wmma16(pa[kk], bv[kk], o[nt]);
        }
        buf ^= 1;
    }

    // finalize: divide by l_i, store f16 [t][512]
#pragma unroll
    for (int v = 0; v < 8; ++v) {
        float inv = 1.0f / l_i[v];
        int t = t0 + m0 + v + 8 * hi;
#pragma unroll
        for (int nt = 0; nt < 4; ++nt)
            ah[(size_t)t * HID + head * 64 + nt * 16 + lo] = (_Float16)(o[nt][v] * inv);
    }
}

// ---------------- output projection: out = ah . wo^T (f32 out) ----------------
// grid (T/128, 8); block 256 (8 waves). A-fragments straight from global (f16, aligned).
__global__ __launch_bounds__(256) void out_proj_kernel(
    const _Float16* __restrict__ ah, const _Float16* __restrict__ woh,
    float* __restrict__ out)
{
    const int t0 = blockIdx.x * 128;
    const int n0 = blockIdx.y * 64;
    const int tid = threadIdx.x;
    const int l = tid & 31, w = tid >> 5;
    const int lo = l & 15, hi = l >> 4;
    const int m0 = w * 16;

    v8f acc[4];
#pragma unroll
    for (int nt = 0; nt < 4; ++nt) acc[nt] = zero8();

    const _Float16* arowBase = ah + (size_t)(t0 + m0 + lo) * HID;
    for (int kc = 0; kc < HID; kc += 32) {
        const _Float16* arow = arowBase + kc + hi * 8;
        v16h a = mk16(*(const half8*)arow, *(const half8*)(arow + 16));
        v16h b[4];
#pragma unroll
        for (int nt = 0; nt < 4; ++nt) {
            const _Float16* brow = woh + (size_t)(n0 + nt * 16 + lo) * HID + kc + hi * 8;
            b[nt] = mk16(*(const half8*)brow, *(const half8*)(brow + 16));
        }
#pragma unroll
        for (int nt = 0; nt < 4; ++nt)
            acc[nt] = wmma16(a, b[nt], acc[nt]);
    }
#pragma unroll
    for (int nt = 0; nt < 4; ++nt)
#pragma unroll
        for (int v = 0; v < 8; ++v) {
            int t = t0 + m0 + v + 8 * hi;
            out[(size_t)t * HID + n0 + nt * 16 + lo] = acc[nt][v];
        }
}

// ---------------- launch ----------------

extern "C" void kernel_launch(void* const* d_in, const int* in_sizes, int n_in,
                              void* d_out, int out_size, void* d_ws, size_t ws_size,
                              hipStream_t stream) {
    const float* x  = (const float*)d_in[0];
    const float* wq = (const float*)d_in[1];
    const float* wk = (const float*)d_in[2];
    const float* wv = (const float*)d_in[3];
    const float* wo = (const float*)d_in[4];
    const int*   ts = (const int*)d_in[5];
    float* out = (float*)d_out;

    char* ws = (char*)d_ws;
    size_t off = 0;
    auto alloc = [&](size_t bytes) -> void* {
        void* p = ws + off;
        off += (bytes + 255) & ~(size_t)255;
        return p;
    };
    _Float16* wqh  = (_Float16*)alloc((size_t)HID * HID * 2);
    _Float16* wkh  = (_Float16*)alloc((size_t)HID * HID * 2);
    _Float16* wvh  = (_Float16*)alloc((size_t)HID * HID * 2);
    _Float16* woh  = (_Float16*)alloc((size_t)HID * HID * 2);
    _Float16* qh   = (_Float16*)alloc((size_t)NHEAD * TSEQ * HDIM * 2);
    _Float16* kh   = (_Float16*)alloc((size_t)NHEAD * TSEQ * HDIM * 2);
    _Float16* vt   = (_Float16*)alloc((size_t)NHEAD * TSEQ * HDIM * 2);  // transposed [h][d][t]
    _Float16* ahbf = (_Float16*)alloc((size_t)TSEQ * HID * 2);
    float*    cosT = (float*)alloc((size_t)TSEQ * 32 * 4);
    float*    sinT = (float*)alloc((size_t)TSEQ * 32 * 4);

    const int wn = HID * HID;
    cvt_f32_to_f16_kernel<<<(wn + 255) / 256, 256, 0, stream>>>(wq, wqh, wn);
    cvt_f32_to_f16_kernel<<<(wn + 255) / 256, 256, 0, stream>>>(wk, wkh, wn);
    cvt_f32_to_f16_kernel<<<(wn + 255) / 256, 256, 0, stream>>>(wv, wvh, wn);
    cvt_f32_to_f16_kernel<<<(wn + 255) / 256, 256, 0, stream>>>(wo, woh, wn);
    rope_table_kernel<<<(TSEQ * 32) / 256, 256, 0, stream>>>(ts, cosT, sinT);

    qkv_rope_kernel<<<dim3(TSEQ / 128, NHEAD, 3), 256, 0, stream>>>(
        x, wqh, wkh, wvh, cosT, sinT, qh, kh, vt);

    flash_attn_kernel<<<dim3(TSEQ / 64, NHEAD), 128, 0, stream>>>(qh, kh, vt, ahbf);

    out_proj_kernel<<<dim3(TSEQ / 128, NHEAD), 256, 0, stream>>>(ahbf, woh, out);
}
